// Transformer_69913477644912
// MI455X (gfx1250) — matrix operand order repaired and checked by reading, not compiled, not run on hardware
//
#include <hip/hip_runtime.h>
#include <stdint.h>

// ---------------------------------------------------------------------------
// GPT-2 small forward (B=2,S=1024,D=768,H=12,L=4,V=50257) for gfx1250 (CDNA5)
// - all GEMMs + attention on v_wmma_f32_16x16x32_bf16 (wave32 WMMA)
// - GEMM tiles staged with GLOBAL_LOAD_ASYNC_TO_LDS_B128 (ASYNCcnt) when the
//   toolchain exposes the gfx1250 async builtins, double-buffered in LDS
// - weights pre-transposed to [N,K] bf16 so all tile copies are K-contiguous
// ---------------------------------------------------------------------------

typedef __bf16 bf16;
typedef __attribute__((ext_vector_type(16))) __bf16 v16bf;
typedef __attribute__((ext_vector_type(8)))  __bf16 v8bf;
typedef __attribute__((ext_vector_type(8)))  float  v8f;

union FragB { v16bf v; v8bf h8[2]; };

#if defined(__gfx1250__) && __has_builtin(__builtin_amdgcn_global_load_async_to_lds_b128)
#define HAVE_ASYNC_LDS 1
#else
#define HAVE_ASYNC_LDS 0
#endif

// async-copy pointer types: (int4 AS1*, int4 AS3*, imm offset, imm cpol)
typedef int i32x4 __attribute__((vector_size(16)));
#if HAVE_ASYNC_LDS
typedef __attribute__((address_space(1))) i32x4* as1_i32x4p;
typedef __attribute__((address_space(3))) i32x4* as3_i32x4p;
#endif

static __device__ __forceinline__ bf16 tobf(float f) {
  union { float f; uint32_t u; } c; c.f = f;
  uint32_t r = c.u + 0x7FFFu + ((c.u >> 16) & 1u);   // RNE
  return __builtin_bit_cast(bf16, (uint16_t)(r >> 16));
}

static __device__ __forceinline__ float gelu_f(float x) {
  const float k = 0.7978845608028654f; // sqrt(2/pi)
  float t = tanhf(k * (x + 0.044715f * x * x * x));
  return 0.5f * x * (1.0f + t);
}

// ---------------------------------------------------------------------------
// f32 -> bf16 (plain)
// ---------------------------------------------------------------------------
__global__ __launch_bounds__(256)
void cvt_kernel(const float* __restrict__ in, bf16* __restrict__ out, long n) {
  long i = (long)blockIdx.x * blockDim.x + threadIdx.x;
  long stride = (long)gridDim.x * blockDim.x;
  for (; i < n; i += stride) out[i] = tobf(in[i]);
}

// ---------------------------------------------------------------------------
// f32 [L,K,N] -> bf16 [L,N,K] (transpose per layer, once per launch)
// ---------------------------------------------------------------------------
__global__ __launch_bounds__(256)
void cvt_t_kernel(const float* __restrict__ in, bf16* __restrict__ out,
                  int LL, int K, int N) {
  long total = (long)LL * K * N;
  long i = (long)blockIdx.x * blockDim.x + threadIdx.x;
  long stride = (long)gridDim.x * blockDim.x;
  for (; i < total; i += stride) {
    long l = i / ((long)N * K);
    long rem = i - l * (long)N * K;
    long n = rem / K;
    long k = rem - n * K;
    out[i] = tobf(in[(l * (long)K + k) * N + n]);
  }
}

// ---------------------------------------------------------------------------
// h[b,s,:] = wte[X[b,s],:] + wpe[s,:]
// ---------------------------------------------------------------------------
__global__ __launch_bounds__(256)
void embed_kernel(const int* __restrict__ X, const float* __restrict__ wte,
                  const float* __restrict__ wpe, float* __restrict__ h) {
  int row = blockIdx.x;            // b*1024 + s
  int s   = row & 1023;
  long tok = (long)X[row];
  float* hr = h + (long)row * 768;
  const float* er = wte + tok * 768;
  const float* pr = wpe + (long)s * 768;
  for (int d = threadIdx.x; d < 768; d += 256) hr[d] = er[d] + pr[d];
}

// ---------------------------------------------------------------------------
// layernorm D=768, f32 in -> bf16 out; one 8-wave block per row
// ---------------------------------------------------------------------------
__global__ __launch_bounds__(256)
void ln_kernel(const float* __restrict__ x, const float* __restrict__ g,
               const float* __restrict__ b, bf16* __restrict__ out) {
  int row = blockIdx.x;
  int t = threadIdx.x;
  const float* xr = x + (long)row * 768;
  float v0 = xr[t], v1 = xr[t + 256], v2 = xr[t + 512];
  float s  = v0 + v1 + v2;
  float ss = v0 * v0 + v1 * v1 + v2 * v2;
#pragma unroll
  for (int m = 16; m >= 1; m >>= 1) {
    s  += __shfl_xor(s,  m, 32);
    ss += __shfl_xor(ss, m, 32);
  }
  __shared__ float r1[8], r2[8];
  if ((t & 31) == 0) { r1[t >> 5] = s; r2[t >> 5] = ss; }
  __syncthreads();
  s = 0.f; ss = 0.f;
#pragma unroll
  for (int i = 0; i < 8; i++) { s += r1[i]; ss += r2[i]; }
  float mean = s * (1.0f / 768.0f);
  float var  = ss * (1.0f / 768.0f) - mean * mean;
  float rs   = rsqrtf(var + 1e-5f);
  bf16* orow = out + (long)row * 768;
  orow[t]       = tobf((v0 - mean) * rs * g[t]       + b[t]);
  orow[t + 256] = tobf((v1 - mean) * rs * g[t + 256] + b[t + 256]);
  orow[t + 512] = tobf((v2 - mean) * rs * g[t + 512] + b[t + 512]);
}

// ---------------------------------------------------------------------------
// WMMA bf16 GEMM: C[M,N] = A[M,K] * B[N,K]^T + bias (+gelu) (+residual)
//   A: bf16 row-major [M,K]  (M%128==0, K%32==0)
//   B: bf16 row-major [N,K]  (ragged N handled by clamp + store guard)
//   out: f32 (outf) or bf16 (outb)
// 256 threads = 8 waves, tile 128x128, k-step 32, wave grid 2(M) x 4(N),
// each wave 64x32 = 4x2 WMMA tiles. Tiles double-buffered in LDS, staged via
// GLOBAL_LOAD_ASYNC_TO_LDS_B128 when available.
// ---------------------------------------------------------------------------
#define BM 128
#define BN 128
#define BK 32
#define LDK (BK + 8)   // 40 bf16 = 80B rows: 16B-aligned b128 fragment reads

__global__ __launch_bounds__(256)
void gemm_bf16_wmma(const bf16* __restrict__ A, const bf16* __restrict__ B,
                    const float* __restrict__ bias, const float* __restrict__ residual,
                    float* __restrict__ outf, bf16* __restrict__ outb,
                    int M, int N, int K, int act) {
  __shared__ __align__(16) bf16 lds_a[2][BM][LDK];
  __shared__ __align__(16) bf16 lds_b[2][BN][LDK];

  const int tid  = threadIdx.x;
  const int lane = tid & 31;
  const int wid  = tid >> 5;
  const int wm   = wid >> 2;        // 0..1
  const int wn   = wid & 3;         // 0..3
  const int bm0  = blockIdx.y * BM;
  const int bn0  = blockIdx.x * BN;

  const int row16 = lane & 15;
  const int kbA = (lane < 16) ? 0 : 16;   // A frag: K = kbA..+15
  const int kbB = (lane < 16) ? 0 : 16;   // B frag: K = kbB..+15

  v8f acc[4][2];
#pragma unroll
  for (int i = 0; i < 4; i++)
#pragma unroll
    for (int j = 0; j < 2; j++)
#pragma unroll
      for (int e = 0; e < 8; e++) acc[i][j][e] = 0.0f;

  // stage one 128x32 tile of A and of B into LDS buffer `buf`
  auto stage = [&](int buf, int k0) {
#pragma unroll
    for (int i = 0; i < 2; i++) {
      int idx = tid + i * 256;          // 0..511
      int r   = idx >> 2;               // 0..127
      int seg = (idx & 3) * 8;          // bf16 elems: 0,8,16,24 (16B chunks)
      int gn  = bn0 + r; if (gn > N - 1) gn = N - 1;   // clamp ragged N
      const bf16* ga = A + (long)(bm0 + r) * K + k0 + seg;
      const bf16* gb = B + (long)gn * K + k0 + seg;
#if HAVE_ASYNC_LDS
      __builtin_amdgcn_global_load_async_to_lds_b128(
          (as1_i32x4p)(void*)ga,
          (as3_i32x4p)(void*)&lds_a[buf][r][seg], 0, 0);
      __builtin_amdgcn_global_load_async_to_lds_b128(
          (as1_i32x4p)(void*)gb,
          (as3_i32x4p)(void*)&lds_b[buf][r][seg], 0, 0);
#else
      *(uint64_t*)&lds_a[buf][r][seg]     = *(const uint64_t*)ga;
      *(uint64_t*)&lds_a[buf][r][seg + 4] = *(const uint64_t*)(ga + 4);
      *(uint64_t*)&lds_b[buf][r][seg]     = *(const uint64_t*)gb;
      *(uint64_t*)&lds_b[buf][r][seg + 4] = *(const uint64_t*)(gb + 4);
#endif
    }
  };

#if HAVE_ASYNC_LDS
#define WAIT_ASYNC() asm volatile("s_wait_asynccnt 0x0" ::: "memory")
#else
#define WAIT_ASYNC()
#endif

  const int nk = K / BK;
  stage(0, 0);
  WAIT_ASYNC();
  __syncthreads();

  for (int kt = 0; kt < nk; ++kt) {
    const int buf = kt & 1;
    if (kt + 1 < nk) stage(buf ^ 1, (kt + 1) * BK);   // overlap with compute

    FragB bfr[2];
#pragma unroll
    for (int nt = 0; nt < 2; nt++) {
      int n = wn * 32 + nt * 16 + row16;
      bfr[nt].h8[0] = *(const v8bf*)&lds_b[buf][n][kbB];
      bfr[nt].h8[1] = *(const v8bf*)&lds_b[buf][n][kbB + 8];
    }
#pragma unroll
    for (int mt = 0; mt < 4; mt++) {
      FragB afr;
      int r = wm * 64 + mt * 16 + row16;
      afr.h8[0] = *(const v8bf*)&lds_a[buf][r][kbA];
      afr.h8[1] = *(const v8bf*)&lds_a[buf][r][kbA + 8];
#pragma unroll
      for (int nt = 0; nt < 2; nt++) {
        acc[mt][nt] = __builtin_amdgcn_wmma_f32_16x16x32_bf16(
            false, afr.v, false, bfr[nt].v, (short)0, acc[mt][nt], false, false);
      }
    }
    if (kt + 1 < nk) WAIT_ASYNC();
    __syncthreads();
  }

  // epilogue: C layout: col = lane&15, rows j + 8*(lane>=16)
  const int colb = lane & 15;
  const int rowb = (lane < 16) ? 0 : 8;
#pragma unroll
  for (int mt = 0; mt < 4; mt++) {
#pragma unroll
    for (int nt = 0; nt < 2; nt++) {
      int gc = bn0 + wn * 32 + nt * 16 + colb;
      if (gc < N) {
        float bv = bias ? bias[gc] : 0.0f;
#pragma unroll
        for (int j = 0; j < 8; j++) {
          int gr = bm0 + wm * 64 + mt * 16 + rowb + j;
          float v = acc[mt][nt][j] + bv;
          if (act) v = gelu_f(v);
          long idx = (long)gr * N + gc;
          if (residual) v += residual[idx];
          if (outb) outb[idx] = tobf(v);
          else      outf[idx] = v;
        }
      }
    }
  }
}

// ---------------------------------------------------------------------------
// Flash attention (causal), H=12, HD=64, S=1024.
// qkv: bf16 [B*S, 2304] (q|k|v, 64 per head); out: bf16 [B*S, 768]
// 128 threads = 4 waves, one 16-query tile per wave; keys streamed 32 at a
// time with online softmax. QK^T and P*V on WMMA; P redistributed through
// per-wave LDS with a same-wave s_wait_dscnt (no barrier needed).
// ---------------------------------------------------------------------------
__global__ __launch_bounds__(128)
void attn_kernel(const bf16* __restrict__ qkv, bf16* __restrict__ out) {
  const int S = 1024, H = 12;
  __shared__ __align__(16) bf16 pst[4][16][40];

  const int tid  = threadIdx.x;
  const int lane = tid & 31;
  const int wave = tid >> 5;
  const int qblocks = S / 64;
  const int qb = blockIdx.x % qblocks;
  const int bh = blockIdx.x / qblocks;
  const int hh = bh % H;
  const int bb = bh / H;
  const int q0 = qb * 64 + wave * 16;

  const int row16 = lane & 15;
  const int colb  = lane & 15;
  const int rowb  = (lane < 16) ? 0 : 8;
  const int kbA   = (lane < 16) ? 0 : 16;
  const int kbB   = (lane < 16) ? 0 : 16;

  const bf16* base = qkv + (long)bb * S * 2304;
  const int qoff = hh * 64, koff = 768 + hh * 64, voff = 1536 + hh * 64;

  // Q fragments (A layout, 16x64 as two 16x32), 1/sqrt(64) folded in
  FragB qa0, qa1;
  {
    const float sc = 0.125f;
    const bf16* qp = base + (long)(q0 + row16) * 2304 + qoff;
#pragma unroll
    for (int j = 0; j < 8; j++) {
      qa0.v[j]     = tobf((float)qp[kbA + j] * sc);
      qa0.v[8 + j] = tobf((float)qp[kbA + 8 + j] * sc);
      qa1.v[j]     = tobf((float)qp[32 + kbA + j] * sc);
      qa1.v[8 + j] = tobf((float)qp[40 + kbA + j] * sc);
    }
  }

  float m_i[8], l_i[8];
  v8f o[4];
#pragma unroll
  for (int j = 0; j < 8; j++) { m_i[j] = -1e30f; l_i[j] = 0.0f; }
#pragma unroll
  for (int t = 0; t < 4; t++)
#pragma unroll
    for (int e = 0; e < 8; e++) o[t][e] = 0.0f;

  // 16x16 score tile vs keys [keybase, keybase+16): K frag = 4x b128 global
  auto score_tile = [&](int keybase) -> v8f {
    const bf16* kp = base + (long)(keybase + colb) * 2304 + koff;
    FragB k0f, k1f;
    k0f.h8[0] = *(const v8bf*)(kp + kbB);
    k0f.h8[1] = *(const v8bf*)(kp + kbB + 8);
    k1f.h8[0] = *(const v8bf*)(kp + 32 + kbB);
    k1f.h8[1] = *(const v8bf*)(kp + 32 + kbB + 8);
    v8f s;
#pragma unroll
    for (int e = 0; e < 8; e++) s[e] = 0.0f;
    s = __builtin_amdgcn_wmma_f32_16x16x32_bf16(false, qa0.v, false, k0f.v,
                                                (short)0, s, false, false);
    s = __builtin_amdgcn_wmma_f32_16x16x32_bf16(false, qa1.v, false, k1f.v,
                                                (short)0, s, false, false);
    return s;
  };

  const int nkv = (q0 + 16 + 31) / 32;
  for (int it = 0; it < nkv; ++it) {
    const int kstart = it * 32;
    v8f sA = score_tile(kstart);
    v8f sB = score_tile(kstart + 16);

#pragma unroll
    for (int j = 0; j < 8; j++) {
      int qi = q0 + rowb + j;
      if (kstart + colb > qi)      sA[j] = -1e30f;
      if (kstart + 16 + colb > qi) sB[j] = -1e30f;

      float mx = fmaxf(sA[j], sB[j]);
#pragma unroll
      for (int m = 8; m >= 1; m >>= 1) mx = fmaxf(mx, __shfl_xor(mx, m, 32));
      float mnew = fmaxf(m_i[j], mx);
      float scl  = __expf(m_i[j] - mnew);
      float pa   = __expf(sA[j] - mnew);
      float pb   = __expf(sB[j] - mnew);
      float rs   = pa + pb;
#pragma unroll
      for (int m = 8; m >= 1; m >>= 1) rs += __shfl_xor(rs, m, 32);
      l_i[j] = l_i[j] * scl + rs;
      m_i[j] = mnew;
#pragma unroll
      for (int t = 0; t < 4; t++) o[t][j] *= scl;
      pst[wave][rowb + j][colb]      = tobf(pa);
      pst[wave][rowb + j][16 + colb] = tobf(pb);
    }
    asm volatile("s_wait_dscnt 0" ::: "memory");   // same-wave LDS RAW

    FragB pf;
    pf.h8[0] = *(const v8bf*)&pst[wave][row16][kbA];
    pf.h8[1] = *(const v8bf*)&pst[wave][row16][kbA + 8];

#pragma unroll
    for (int t = 0; t < 4; t++) {
      FragB vf;
#pragma unroll
      for (int j = 0; j < 16; j++) {
        vf.v[j] = base[(long)(kstart + kbB + j) * 2304 + voff + t * 16 + colb];
      }
      o[t] = __builtin_amdgcn_wmma_f32_16x16x32_bf16(false, pf.v, false, vf.v,
                                                     (short)0, o[t], false, false);
    }
  }

#pragma unroll
  for (int t = 0; t < 4; t++) {
#pragma unroll
    for (int j = 0; j < 8; j++) {
      float v = o[t][j] / l_i[j];
      long idx = (long)(bb * S + q0 + rowb + j) * 768 + hh * 64 + t * 16 + colb;
      out[idx] = tobf(v);
    }
  }
}

// ---------------------------------------------------------------------------
// launch
// ---------------------------------------------------------------------------
extern "C" void kernel_launch(void* const* d_in, const int* in_sizes, int n_in,
                              void* d_out, int out_size, void* d_ws, size_t ws_size,
                              hipStream_t stream) {
  (void)in_sizes; (void)n_in; (void)out_size; (void)ws_size;
  const int Bq = 2, S = 1024, D = 768, H = 12, L = 4, V = 50257;
  const int M = Bq * S;  // 2048

  const int*   X      = (const int*)  d_in[0];
  const float* wte    = (const float*)d_in[1];
  const float* wpe    = (const float*)d_in[2];
  const float* ln1_g  = (const float*)d_in[3];
  const float* ln1_b  = (const float*)d_in[4];
  const float* attn_w = (const float*)d_in[5];
  const float* attn_b = (const float*)d_in[6];
  const float* proj_w = (const float*)d_in[7];
  const float* proj_b = (const float*)d_in[8];
  const float* ln2_g  = (const float*)d_in[9];
  const float* ln2_b  = (const float*)d_in[10];
  const float* fc_w   = (const float*)d_in[11];
  const float* fc_b   = (const float*)d_in[12];
  const float* fc2_w  = (const float*)d_in[13];
  const float* fc2_b  = (const float*)d_in[14];
  const float* lnf_g  = (const float*)d_in[15];
  const float* lnf_b  = (const float*)d_in[16];

  char* ws = (char*)d_ws;
  size_t off = 0;
  auto carve = [&](size_t bytes) -> char* {
    char* p = ws + off;
    off += (bytes + 255) & ~(size_t)255;
    return p;
  };
  // weights, bf16, transposed to [N,K] per layer (wte is already [V,D]=[N,K])
  bf16* wteb    = (bf16*) carve((size_t)V * D * 2);
  bf16* attnwt  = (bf16*) carve((size_t)L * 3 * D * D * 2);
  bf16* projwt  = (bf16*) carve((size_t)L * D * D * 2);
  bf16* fcwt    = (bf16*) carve((size_t)L * 4 * D * D * 2);
  bf16* fc2wt   = (bf16*) carve((size_t)L * D * 4 * D * 2);
  float* hbuf   = (float*)carve((size_t)M * D * 4);
  bf16* lnbuf   = (bf16*) carve((size_t)M * D * 2);
  bf16* qkvb    = (bf16*) carve((size_t)M * 3 * D * 2);
  bf16* attnout = (bf16*) carve((size_t)M * D * 2);
  bf16* midbuf  = (bf16*) carve((size_t)M * 4 * D * 2);

  cvt_kernel  <<<4096, 256, 0, stream>>>(wte, wteb, (long)V * D);
  cvt_t_kernel<<<4096, 256, 0, stream>>>(attn_w, attnwt, L, D, 3 * D);
  cvt_t_kernel<<<2048, 256, 0, stream>>>(proj_w, projwt, L, D, D);
  cvt_t_kernel<<<4096, 256, 0, stream>>>(fc_w,  fcwt,  L, D, 4 * D);
  cvt_t_kernel<<<4096, 256, 0, stream>>>(fc2_w, fc2wt, L, 4 * D, D);

  embed_kernel<<<M, 256, 0, stream>>>(X, wte, wpe, hbuf);

  const dim3 blk(256);
  for (int l = 0; l < L; l++) {
    ln_kernel<<<M, 256, 0, stream>>>(hbuf, ln1_g + l * D, ln1_b + l * D, lnbuf);
    // qkv (bf16 out) = LN1(h) @ attn_w + attn_b   [2048 x 2304]
    gemm_bf16_wmma<<<dim3(3 * D / BN, M / BM), blk, 0, stream>>>(
        lnbuf, attnwt + (size_t)l * 3 * D * D, attn_b + l * 3 * D,
        nullptr, nullptr, qkvb, M, 3 * D, D, 0);
    attn_kernel<<<Bq * H * (S / 64), 128, 0, stream>>>(qkvb, attnout);
    // h += attnout @ proj_w + proj_b
    gemm_bf16_wmma<<<dim3(D / BN, M / BM), blk, 0, stream>>>(
        attnout, projwt + (size_t)l * D * D, proj_b + l * D,
        hbuf, hbuf, nullptr, M, D, D, 0);
    ln_kernel<<<M, 256, 0, stream>>>(hbuf, ln2_g + l * D, ln2_b + l * D, lnbuf);
    // mid (bf16) = gelu(LN2(h) @ fc_w + fc_b)   [2048 x 3072]
    gemm_bf16_wmma<<<dim3(4 * D / BN, M / BM), blk, 0, stream>>>(
        lnbuf, fcwt + (size_t)l * 4 * D * D, fc_b + l * 4 * D,
        nullptr, nullptr, midbuf, M, 4 * D, D, 1);
    // h += mid @ fc2_w + fc2_b
    gemm_bf16_wmma<<<dim3(D / BN, M / BM), blk, 0, stream>>>(
        midbuf, fc2wt + (size_t)l * D * 4 * D, fc2_b + l * D,
        hbuf, hbuf, nullptr, M, D, 4 * D, 0);
  }

  ln_kernel<<<M, 256, 0, stream>>>(hbuf, lnf_g, lnf_b, lnbuf);
  // logits = LNf(h) @ wte^T   [2048 x 50257]
  gemm_bf16_wmma<<<dim3((V + BN - 1) / BN, M / BM), blk, 0, stream>>>(
      lnbuf, wteb, nullptr, nullptr, (float*)d_out, nullptr, M, V, D, 0);
}
